// VAE_47562467836382
// MI455X (gfx1250) — compile-verified, matrix operand order checked by reading, not run on hardware
//
#include <hip/hip_runtime.h>
#include <hip/hip_bf16.h>
#include <math.h>

// ---------------- problem dims ----------------
#define TT   8192      // frames
#define LDIM 442       // linguistic dim
#define ADIM 199       // acoustic dim
#define HDIM 400       // lstm hidden
#define G4H  1600      // 4*H gate width
#define ZDIM 4
#define NMOR 512

// padded K dims (multiples of 32 for wmma f16 K-chunks)
#define K1P 672        // pad(442+199=641)
#define K2P 448        // pad(442)  (z columns added separately in f32)
#define KHP 416        // pad(400)
#define KFC 800        // 2H, already multiple of 32
#define NFC 208        // pad(199) to multiple of 16

typedef __attribute__((ext_vector_type(16))) _Float16 v16h;
typedef __attribute__((ext_vector_type(8)))  _Float16 v8h;
typedef __attribute__((ext_vector_type(8)))  float    v8f;

// global-address-space (AS1) vector pointer: guarantees global_load_b128
// (not flat_load_b128, which would also tie up DScnt)
typedef const __attribute__((address_space(1))) v8h* gv8hp;

__device__ __forceinline__ v8h load_v8h_g(const _Float16* p) {
  return *(gv8hp)(unsigned long long)p;
}

// Load a 16-f16 wmma fragment from GLOBAL memory. Caller passes the
// lane-adjusted base (row base + K-half offset); elements {0..7} contiguous,
// {8..15} live +16 elements further (K pattern {0..7,16..23}/{8..15,24..31}).
__device__ __forceinline__ v16h load_frag16_g(const _Float16* p) {
  gv8hp gp = (gv8hp)(unsigned long long)p;
  v8h lo = gp[0];
  v8h hi = gp[2];                        // +16 halves = +2 v8h
  return __builtin_shufflevector(lo, hi, 0,1,2,3,4,5,6,7,8,9,10,11,12,13,14,15);
}

// Same but through the generic path (used for LDS)
__device__ __forceinline__ v16h load_frag16(const _Float16* p) {
  v8h lo = *(const v8h*)(p);
  v8h hi = *(const v8h*)(p + 16);
  return __builtin_shufflevector(lo, hi, 0,1,2,3,4,5,6,7,8,9,10,11,12,13,14,15);
}

// ---------------- staging kernels ----------------
__global__ void zero_f16_kernel(_Float16* dst, size_t n) {
  size_t i = (size_t)blockIdx.x * blockDim.x + threadIdx.x;
  if (i < n) dst[i] = (_Float16)0.f;
}

// dst[r*dstLd + dstColOff + c] = (f16) src[r*srcLd + c], r<rows, c<cols
__global__ void pack_f16_kernel(_Float16* dst, int dstLd, int dstColOff,
                                const float* __restrict__ src, int srcLd,
                                int rows, int cols) {
  int idx = blockIdx.x * blockDim.x + threadIdx.x;
  if (idx >= rows * cols) return;
  int r = idx / cols, c = idx % cols;
  dst[(size_t)r * dstLd + dstColOff + c] = (_Float16)src[(size_t)r * srcLd + c];
}

// ---------------- WMMA GEMM:  out[m,n] = sum_k A[m,k]*W[n,k] + bias ----------------
// A: [M x lda] f16 row-major, W: [N x ldb] f16 row-major (row n = weights of out col n)
template<bool OUTF16>
__global__ void gemm_wmma_nt(const _Float16* __restrict__ A, int lda,
                             const _Float16* __restrict__ W, int ldb,
                             const float* __restrict__ bias1,
                             const float* __restrict__ bias2,
                             float* __restrict__ outF,
                             _Float16* __restrict__ outH, int ldo,
                             int M, int N, int Nvalid, int K) {
  const int wave = threadIdx.x >> 5;
  const int lane = threadIdx.x & 31;
  const int tilesN = N >> 4;
  const int tile = blockIdx.x * (blockDim.x >> 5) + wave;
  if (tile >= (M >> 4) * tilesN) return;
  const int tm = tile / tilesN, tn = tile % tilesN;
  const int hs = (lane < 16) ? 0 : 8;           // K-half select per layout
  const _Float16* ap = A + (size_t)(tm * 16 + (lane & 15)) * lda + hs;
  const _Float16* wp = W + (size_t)(tn * 16 + (lane & 15)) * ldb + hs;
  v8f acc = {};
  for (int k0 = 0; k0 < K; k0 += 32) {
    v16h a = load_frag16_g(ap + k0);
    v16h b = load_frag16_g(wp + k0);
    acc = __builtin_amdgcn_wmma_f32_16x16x32_f16(false, a, false, b,
                                                 (short)0, acc, false, false);
  }
  const int col = tn * 16 + (lane & 15);
  if (col < Nvalid) {
    float bv = (bias1 ? bias1[col] : 0.f) + (bias2 ? bias2[col] : 0.f);
#pragma unroll
    for (int r = 0; r < 8; ++r) {
      int m = tm * 16 + r + hs;                  // D layout: lane half -> M+8
      float v = acc[r] + bv;
      if (OUTF16) outH[(size_t)m * ldo + col] = (_Float16)v;
      else        outF[(size_t)m * ldo + col] = v;
    }
  }
}

// ---------------- sequential LSTM scan (one block per direction) ----------------
// Per step: g[1600] = pre[t] + Whh(1600x400) @ h(400) via wmma with h replicated
// across all 16 B columns (every column of D equals the matvec result).
// Whh (1.33MB f16/direction) cannot be register- or LDS-resident, so it is
// deliberately re-streamed from L2 every step: an opaque laundered offset makes
// the weight loads loop-variant (no hoist -> no scratch spills) while keeping
// the pointer's global provenance (global_load_b128, LOADcnt only).
// The 13 h-vector B fragments are preloaded once per step into registers
// (104 VGPRs) and shared by all 4 M-tiles: one DScnt wait per step.
__global__ __launch_bounds__(800)
void lstm_scan_kernel(const _Float16* __restrict__ preF,
                      const _Float16* __restrict__ preB,
                      const _Float16* __restrict__ whhF,
                      const _Float16* __restrict__ whhB,
                      const int* __restrict__ mora_pos,
                      float* __restrict__ hmF, float* __restrict__ hmB,
                      _Float16* __restrict__ h3, int phase) {
  const int dir = blockIdx.x;                    // 0 = fwd, 1 = bwd
  const _Float16* __restrict__ pre = dir ? preB : preF;
  const _Float16* whh0 = dir ? whhB : whhF;
  float* __restrict__ hm = dir ? hmB : hmF;
  const int coloff = dir ? HDIM : 0;

  __shared__ __align__(16) _Float16 hbuf[KHP];   // current h (f16, zero-padded)
  __shared__ float gbuf[G4H];                    // gate pre-activations
  __shared__ float cst[HDIM];                    // cell state
  __shared__ int   mpos[NMOR];

  const int tid = threadIdx.x;
  const int wave = tid >> 5, lane = tid & 31;
  const int hs = (lane < 16) ? 0 : 8;
  if (tid < KHP)  hbuf[tid] = (_Float16)0.f;
  if (tid < HDIM) cst[tid] = 0.f;
  if (tid < NMOR) mpos[tid] = mora_pos[tid];
  __syncthreads();

  const int mt0 = wave * 4;                      // 25 waves x 4 tiles = 100 M-tiles

  for (int step = 0; step < TT; ++step) {
    const int t = dir ? (TT - 1 - step) : step;
    const _Float16* prow = pre + (size_t)t * G4H;

    // opaque zero offset: keeps weight loads loop-variant so they stream from
    // L2 each step instead of being hoisted (and spilled to scratch)
    size_t zoff = 0;
    asm volatile("" : "+s"(zoff));
    const _Float16* whh_s = whh0 + zoff;

    // prefetch next step's pre row (13 x 256B lines) while we compute
    if (wave == 0 && lane < 13 && step + 1 < TT) {
      const int tn2 = dir ? (TT - 2 - step) : (step + 1);
      __builtin_prefetch(pre + (size_t)tn2 * G4H + (size_t)lane * 128, 0, 1);
    }

    // preload all B fragments of h once (shared across the 4 M-tiles)
    v16h bfr[13];
#pragma unroll
    for (int kc = 0; kc < 13; ++kc)
      bfr[kc] = load_frag16(&hbuf[kc * 32 + hs]);  // h replicated per column

#pragma unroll
    for (int mi = 0; mi < 4; ++mi) {
      const int mt = mt0 + mi;
      v8f acc;
      v8h pv = load_v8h_g(prow + mt * 16 + hs);    // C init = pre row (same all N)
#pragma unroll
      for (int r = 0; r < 8; ++r) acc[r] = (float)pv[r];
      const _Float16* wrow = whh_s + (size_t)(mt * 16 + (lane & 15)) * KHP + hs;
      for (int kc = 0; kc < 13; ++kc) {
        v16h a = load_frag16_g(wrow + kc * 32);
        acc = __builtin_amdgcn_wmma_f32_16x16x32_f16(false, a, false, bfr[kc],
                                                     (short)0, acc, false, false);
      }
      if ((lane & 15) == 0) {                    // all N columns equal: lanes 0/16
#pragma unroll
        for (int r = 0; r < 8; ++r) gbuf[mt * 16 + r + hs] = acc[r];
      }
    }
    __syncthreads();
    if (tid < HDIM) {                            // gate math (PyTorch i,f,g,o)
      float gi = gbuf[tid];
      float gf = gbuf[HDIM + tid];
      float gg = gbuf[2 * HDIM + tid];
      float go = gbuf[3 * HDIM + tid];
      float si = 1.f / (1.f + expf(-gi));
      float sf = 1.f / (1.f + expf(-gf));
      float so = 1.f / (1.f + expf(-go));
      float c = sf * cst[tid] + si * tanhf(gg);
      cst[tid] = c;
      float h = so * tanhf(c);
      hbuf[tid] = (_Float16)h;
      if (phase == 1) {                          // store raw h at mora rows only
        int lo = 0, hi = NMOR;
        while (lo < hi) { int mid = (lo + hi) >> 1;
                          if (mpos[mid] <= t) lo = mid + 1; else hi = mid; }
        if (lo > 0 && mpos[lo - 1] == t)
          hm[(size_t)(lo - 1) * HDIM + tid] = h;
      } else {                                   // store relu(h) as fc3 operand
        h3[(size_t)t * (2 * HDIM) + coloff + tid] = (_Float16)fmaxf(h, 0.f);
      }
    }
    __syncthreads();
  }
}

// ---------------- encoder head: mu/logvar = relu(h1) @ fc2x_W^T + b ----------------
__global__ void encoder_head_kernel(const float* __restrict__ hf,
                                    const float* __restrict__ hb,
                                    const float* __restrict__ W21, const float* __restrict__ b21,
                                    const float* __restrict__ W22, const float* __restrict__ b22,
                                    float* __restrict__ muOut, float* __restrict__ lvOut,
                                    float* __restrict__ muWs, float* __restrict__ lvWs) {
  int idx = blockIdx.x * blockDim.x + threadIdx.x;
  if (idx >= NMOR * ZDIM * 2) return;
  int which = idx & 1;                           // 0 = mu, 1 = logvar
  int zi = (idx >> 1) & 3;
  int nm = idx >> 3;
  const float* W = which ? W22 : W21;
  const float* b = which ? b22 : b21;
  const float* wr = W + (size_t)zi * (2 * HDIM);
  const float* hfr = hf + (size_t)nm * HDIM;
  const float* hbr = hb + (size_t)nm * HDIM;
  float s = b[zi];
  for (int d = 0; d < HDIM; ++d) s += fmaxf(hfr[d], 0.f) * wr[d];
  for (int d = 0; d < HDIM; ++d) s += fmaxf(hbr[d], 0.f) * wr[HDIM + d];
  if (which) { lvOut[nm * ZDIM + zi] = s; lvWs[nm * ZDIM + zi] = s; }
  else       { muOut[nm * ZDIM + zi] = s; muWs[nm * ZDIM + zi] = s; }
}

__global__ void reparam_kernel(const float* __restrict__ mu,
                               const float* __restrict__ lv,
                               const float* __restrict__ eps,
                               float* __restrict__ z) {
  int i = blockIdx.x * blockDim.x + threadIdx.x;
  if (i < NMOR * ZDIM) z[i] = mu[i] + eps[i] * expf(0.5f * lv[i]);
}

// ---------------- add Wih2[:,442:446] @ z_tmp[t] onto pre2 (f16 RMW) ----------------
__global__ void z_add_kernel(_Float16* __restrict__ preF, _Float16* __restrict__ preB,
                             const float* __restrict__ WihF, const float* __restrict__ WihB,
                             const float* __restrict__ z, const int* __restrict__ mora_pos) {
  int idx = blockIdx.x * blockDim.x + threadIdx.x;
  if (idx >= TT * G4H) return;
  int t = idx / G4H, j = idx % G4H;
  int lo = 0, hi = NMOR;                         // seg[t] = #(mora_pos <= t)
  while (lo < hi) { int mid = (lo + hi) >> 1;
                    if (mora_pos[mid] <= t) lo = mid + 1; else hi = mid; }
  if (lo >= NMOR) return;                        // z_tmp == 0 past last boundary
  const float* zr = z + (size_t)lo * ZDIM;
  float af = 0.f, ab = 0.f;
#pragma unroll
  for (int q = 0; q < ZDIM; ++q) {
    af += WihF[(size_t)j * (LDIM + ZDIM) + LDIM + q] * zr[q];
    ab += WihB[(size_t)j * (LDIM + ZDIM) + LDIM + q] * zr[q];
  }
  size_t o = (size_t)t * G4H + j;
  preF[o] = (_Float16)((float)preF[o] + af);
  preB[o] = (_Float16)((float)preB[o] + ab);
}

// ---------------- host orchestration ----------------
extern "C" void kernel_launch(void* const* d_in, const int* in_sizes, int n_in,
                              void* d_out, int out_size, void* d_ws, size_t ws_size,
                              hipStream_t stream) {
  (void)in_sizes; (void)n_in; (void)out_size; (void)ws_size;
  // inputs in setup_inputs() dict (insertion) order, params flattened recursively
  const float* ling   = (const float*)d_in[0];
  const float* ac     = (const float*)d_in[1];
  const int*   mpos   = (const int*)  d_in[3];   // mora_positions (mora_index unused)
  const float* eps    = (const float*)d_in[4];
  const float* w1f_ih = (const float*)d_in[5];
  const float* w1f_hh = (const float*)d_in[6];
  const float* b1f_ih = (const float*)d_in[7];
  const float* b1f_hh = (const float*)d_in[8];
  const float* w1b_ih = (const float*)d_in[9];
  const float* w1b_hh = (const float*)d_in[10];
  const float* b1b_ih = (const float*)d_in[11];
  const float* b1b_hh = (const float*)d_in[12];
  const float* fc21W  = (const float*)d_in[13];
  const float* fc21b  = (const float*)d_in[14];
  const float* fc22W  = (const float*)d_in[15];
  const float* fc22b  = (const float*)d_in[16];
  const float* w2f_ih = (const float*)d_in[17];
  const float* w2f_hh = (const float*)d_in[18];
  const float* b2f_ih = (const float*)d_in[19];
  const float* b2f_hh = (const float*)d_in[20];
  const float* w2b_ih = (const float*)d_in[21];
  const float* w2b_hh = (const float*)d_in[22];
  const float* b2b_ih = (const float*)d_in[23];
  const float* b2b_hh = (const float*)d_in[24];
  const float* fc3W   = (const float*)d_in[25];
  const float* fc3b   = (const float*)d_in[26];

  float* out = (float*)d_out;                    // (dec, mu, logvar) flat
  float* dec = out;
  float* muO = out + (size_t)TT * ADIM;
  float* lvO = muO + (size_t)NMOR * ZDIM;

  char* base = (char*)d_ws;
  size_t off = 0;
  auto carve = [&](size_t elems, size_t esz) -> void* {
    void* p = base + off;
    off = (off + elems * esz + 255) & ~(size_t)255;
    return p;
  };
  // ---- f16 region (zero-filled once per call so pads are valid) ----
  _Float16* xcat  = (_Float16*)carve((size_t)TT * K1P, 2);   // [ling||ac] f16
  _Float16* lingp = (_Float16*)carve((size_t)TT * K2P, 2);   // ling f16
  _Float16* w1fh  = (_Float16*)carve((size_t)G4H * K1P, 2);
  _Float16* w1bh  = (_Float16*)carve((size_t)G4H * K1P, 2);
  _Float16* w2fh  = (_Float16*)carve((size_t)G4H * K2P, 2);
  _Float16* w2bh  = (_Float16*)carve((size_t)G4H * K2P, 2);
  _Float16* whh1f = (_Float16*)carve((size_t)G4H * KHP, 2);
  _Float16* whh1b = (_Float16*)carve((size_t)G4H * KHP, 2);
  _Float16* whh2f = (_Float16*)carve((size_t)G4H * KHP, 2);
  _Float16* whh2b = (_Float16*)carve((size_t)G4H * KHP, 2);
  _Float16* wfc3  = (_Float16*)carve((size_t)NFC * KFC, 2);
  _Float16* h3    = (_Float16*)carve((size_t)TT * (2 * HDIM), 2);
  _Float16* preA  = (_Float16*)carve((size_t)TT * G4H, 2);   // pre1f then pre2f
  _Float16* preB  = (_Float16*)carve((size_t)TT * G4H, 2);   // pre1b then pre2b
  size_t f16Bytes = off;
  // ---- f32 region ----
  float* hf1m = (float*)carve((size_t)NMOR * HDIM, 4);
  float* hb1m = (float*)carve((size_t)NMOR * HDIM, 4);
  float* muW  = (float*)carve((size_t)NMOR * ZDIM, 4);
  float* lvW  = (float*)carve((size_t)NMOR * ZDIM, 4);
  float* zbuf = (float*)carve((size_t)NMOR * ZDIM, 4);

  auto cdiv = [](long n, long d) { return (int)((n + d - 1) / d); };

  // 0) zero the whole f16 region (pads must be zero for wmma K/N padding)
  size_t f16n = f16Bytes / 2;
  zero_f16_kernel<<<cdiv((long)f16n, 256), 256, 0, stream>>>((_Float16*)base, f16n);

  // 1) pack f32 -> padded f16 staging
  pack_f16_kernel<<<cdiv((long)TT * LDIM, 256), 256, 0, stream>>>(xcat, K1P, 0, ling, LDIM, TT, LDIM);
  pack_f16_kernel<<<cdiv((long)TT * ADIM, 256), 256, 0, stream>>>(xcat, K1P, LDIM, ac, ADIM, TT, ADIM);
  pack_f16_kernel<<<cdiv((long)TT * LDIM, 256), 256, 0, stream>>>(lingp, K2P, 0, ling, LDIM, TT, LDIM);
  pack_f16_kernel<<<cdiv((long)G4H * 641, 256), 256, 0, stream>>>(w1fh, K1P, 0, w1f_ih, 641, G4H, 641);
  pack_f16_kernel<<<cdiv((long)G4H * 641, 256), 256, 0, stream>>>(w1bh, K1P, 0, w1b_ih, 641, G4H, 641);
  pack_f16_kernel<<<cdiv((long)G4H * LDIM, 256), 256, 0, stream>>>(w2fh, K2P, 0, w2f_ih, LDIM + ZDIM, G4H, LDIM);
  pack_f16_kernel<<<cdiv((long)G4H * LDIM, 256), 256, 0, stream>>>(w2bh, K2P, 0, w2b_ih, LDIM + ZDIM, G4H, LDIM);
  pack_f16_kernel<<<cdiv((long)G4H * HDIM, 256), 256, 0, stream>>>(whh1f, KHP, 0, w1f_hh, HDIM, G4H, HDIM);
  pack_f16_kernel<<<cdiv((long)G4H * HDIM, 256), 256, 0, stream>>>(whh1b, KHP, 0, w1b_hh, HDIM, G4H, HDIM);
  pack_f16_kernel<<<cdiv((long)G4H * HDIM, 256), 256, 0, stream>>>(whh2f, KHP, 0, w2f_hh, HDIM, G4H, HDIM);
  pack_f16_kernel<<<cdiv((long)G4H * HDIM, 256), 256, 0, stream>>>(whh2b, KHP, 0, w2b_hh, HDIM, G4H, HDIM);
  pack_f16_kernel<<<cdiv((long)ADIM * KFC, 256), 256, 0, stream>>>(wfc3, KFC, 0, fc3W, KFC, ADIM, KFC);

  // 2) pre1 = [ling||ac] @ Wih1^T + bih + bhh   (both directions), f16 out
  const int tilesPre = (TT / 16) * (G4H / 16);   // 51200, /8 waves = 6400 blocks
  gemm_wmma_nt<true><<<tilesPre / 8, 256, 0, stream>>>(xcat, K1P, w1fh, K1P, b1f_ih, b1f_hh,
                                                       nullptr, preA, G4H, TT, G4H, G4H, K1P);
  gemm_wmma_nt<true><<<tilesPre / 8, 256, 0, stream>>>(xcat, K1P, w1bh, K1P, b1b_ih, b1b_hh,
                                                       nullptr, preB, G4H, TT, G4H, G4H, K1P);

  // 3) BiLSTM-1 scan; keep raw h only at the 512 mora rows
  lstm_scan_kernel<<<2, 800, 0, stream>>>(preA, preB, whh1f, whh1b, mpos,
                                          hf1m, hb1m, nullptr, 1);

  // 4) encoder head + reparameterize
  encoder_head_kernel<<<cdiv(NMOR * ZDIM * 2, 256), 256, 0, stream>>>(
      hf1m, hb1m, fc21W, fc21b, fc22W, fc22b, muO, lvO, muW, lvW);
  reparam_kernel<<<cdiv(NMOR * ZDIM, 256), 256, 0, stream>>>(muW, lvW, eps, zbuf);

  // 5) pre2 base = ling @ Wih2[:, :442]^T + biases (reuse pre buffers), then +z
  gemm_wmma_nt<true><<<tilesPre / 8, 256, 0, stream>>>(lingp, K2P, w2fh, K2P, b2f_ih, b2f_hh,
                                                       nullptr, preA, G4H, TT, G4H, G4H, K2P);
  gemm_wmma_nt<true><<<tilesPre / 8, 256, 0, stream>>>(lingp, K2P, w2bh, K2P, b2b_ih, b2b_hh,
                                                       nullptr, preB, G4H, TT, G4H, G4H, K2P);
  z_add_kernel<<<cdiv((long)TT * G4H, 256), 256, 0, stream>>>(preA, preB, w2f_ih, w2b_ih,
                                                              zbuf, mpos);

  // 6) BiLSTM-2 scan; emit relu(h) as f16 fc3 operand [T x 800]
  lstm_scan_kernel<<<2, 800, 0, stream>>>(preA, preB, whh2f, whh2b, mpos,
                                          nullptr, nullptr, h3, 2);

  // 7) dec = relu(h3) @ fc3_W^T + fc3_b  (f32 into d_out, N padded 199->208)
  const int tilesFc3 = (TT / 16) * (NFC / 16);   // 6656, /8 waves = 832 blocks
  gemm_wmma_nt<false><<<tilesFc3 / 8, 256, 0, stream>>>(h3, KFC, wfc3, KFC, fc3b, nullptr,
                                                        dec, nullptr, ADIM, TT, NFC, ADIM, KFC);
}
